// SpatiotemporalAttention_24953759989770
// MI455X (gfx1250) — matrix operand order repaired and checked by reading, not compiled, not run on hardware
//
#include <hip/hip_runtime.h>
#include <hip/hip_bf16.h>

typedef __bf16 bf16_t;
typedef __attribute__((ext_vector_type(8)))  __bf16 v8bf;
typedef __attribute__((ext_vector_type(16))) __bf16 v16bf;
typedef __attribute__((ext_vector_type(8)))  float  v8f;

#define TOKENS   98304   // B*S*N = 4*12*2048
#define LTOK     24576   // S*N
#define DMODEL   512
#define NHEADS   8
#define HDIM     64
#define KV_CHUNK 1024
#define NCHUNK   24      // LTOK / KV_CHUNK
#define PART_STRIDE 4160 // 64*64 kv + 64 ksum

#ifndef __has_builtin
#define __has_builtin(x) 0
#endif
#if __has_builtin(__builtin_amdgcn_tensor_load_to_lds)
#define USE_TDM 1
#else
#define USE_TDM 0
#endif

static __device__ __forceinline__ v16bf cat8(v8bf lo, v8bf hi) {
  return __builtin_shufflevector(lo, hi, 0,1,2,3,4,5,6,7,8,9,10,11,12,13,14,15);
}

#if USE_TDM
typedef __attribute__((ext_vector_type(4))) unsigned int v4u;
typedef __attribute__((ext_vector_type(4))) int v4i;
typedef __attribute__((ext_vector_type(8))) int v8i_t;

// 2D tile DMA: global (row-major, stride0 elements of 2 bytes) -> LDS,
// optional LDS padding: after (2<<pad_int) DWORDs written, skip (pad_amt+1) DWORDs.
static __device__ __forceinline__ void tdm_load_2d(
    unsigned lds_off, const void* gptr,
    unsigned tile_d0, unsigned tile_d1,
    unsigned tensor_d0, unsigned tensor_d1, unsigned stride0,
    unsigned pad_int, unsigned pad_amt, unsigned pad_en)
{
  unsigned long long ga = (unsigned long long)(size_t)gptr;
  v4u g0;
  g0.x = 1u;                                  // count=1 (valid user descriptor)
  g0.y = lds_off;                             // D#.lds_addr (bytes)
  g0.z = (unsigned)ga;                        // global_addr[31:0]
  g0.w = (unsigned)((ga >> 32) & 0x01FFFFFFu) | (2u << 30);  // addr[56:32] | type=2
  unsigned flags = (1u << 16);                // data_size=1 -> 2-byte elements
  if (pad_en) flags |= (1u << 20) | (pad_int << 22) | (pad_amt << 25);
  v8i_t g1;
  g1[0] = (int)flags;                                       // wg_mask=0 | flags
  g1[1] = (int)((tensor_d0 & 0xFFFFu) << 16);               // dim0[15:0] @ bits[63:48]
  g1[2] = (int)((tensor_d0 >> 16) | ((tensor_d1 & 0xFFFFu) << 16));
  g1[3] = (int)((tensor_d1 >> 16) | (tile_d0 << 16));       // dim1 hi | tile_dim0
  g1[4] = (int)(tile_d1 & 0xFFFFu);                         // tile_dim1 | tile_dim2=0
  g1[5] = (int)stride0;                                     // dim0_stride[31:0]
  g1[6] = 0;
  g1[7] = 0;
  v4i g2 = {0, 0, 0, 0};
  v4i g3 = {0, 0, 0, 0};
#if __clang_major__ >= 23
  v8i_t g4 = {0, 0, 0, 0, 0, 0, 0, 0};
  __builtin_amdgcn_tensor_load_to_lds(g0, g1, g2, g3, g4, 0);
#else
  __builtin_amdgcn_tensor_load_to_lds(g0, g1, g2, g3, 0);
#endif
}
#endif // USE_TDM

// ---------------------------------------------------------------- LayerNorm
__global__ __launch_bounds__(256)
void ln_kernel(const float* __restrict__ x, const float* __restrict__ g,
               const float* __restrict__ bb, bf16_t* __restrict__ h)
{
  const size_t base = (size_t)blockIdx.x * DMODEL;
  const int tid = threadIdx.x;
  float a0 = x[base + tid];
  float a1 = x[base + tid + 256];
  __shared__ float rs[256], rq[256];
  rs[tid] = a0 + a1;
  rq[tid] = a0 * a0 + a1 * a1;
  __syncthreads();
  for (int o = 128; o > 0; o >>= 1) {
    if (tid < o) { rs[tid] += rs[tid + o]; rq[tid] += rq[tid + o]; }
    __syncthreads();
  }
  const float mu   = rs[0] * (1.0f / DMODEL);
  const float var  = rq[0] * (1.0f / DMODEL) - mu * mu;
  const float rstd = rsqrtf(var + 1e-5f);
  h[base + tid]       = (bf16_t)((a0 - mu) * rstd * g[tid] + bb[tid]);
  h[base + tid + 256] = (bf16_t)((a1 - mu) * rstd * g[tid + 256] + bb[tid + 256]);
}

// ---------------------------------------------------------------- fp32 -> bf16
__global__ __launch_bounds__(256)
void cvt_bf16(const float* __restrict__ src, bf16_t* __restrict__ dst, int n)
{
  int i = blockIdx.x * 256 + threadIdx.x;
  if (i < n) dst[i] = (bf16_t)src[i];
}

// ---------------------------------------------------------------- GEMM: out[t,j] = sum_d A[t,d]*W[j,d] + bias[j]
// mode 0: store bf16; mode 1: elu+1 then store bf16; mode 2: + residual, store fp32
__global__ __launch_bounds__(256)
void gemm_bf16(const bf16_t* __restrict__ A, const bf16_t* __restrict__ W,
               const float* __restrict__ bias, const float* __restrict__ resid,
               bf16_t* __restrict__ outb, float* __restrict__ outf, int mode)
{
  __shared__ bf16_t sA[2][128 * 40];   // pitch 40 bf16 (64B row + 16B pad)
  __shared__ bf16_t sB[2][64 * 40];
  const int tid  = threadIdx.x;
  const int lane = tid & 31, wave = tid >> 5;
  const int wm = wave >> 1, wn = wave & 1;
  const long m0 = (long)blockIdx.x * 128;
  const int  n0 = blockIdx.y * 64;

  v8f acc[2][2] = {};

  if (mode == 2 && tid < 128)
    __builtin_prefetch(resid + (m0 + tid) * DMODEL + n0, 0, 1);

#if USE_TDM
  auto issueTiles = [&](int k0, int buf) {
    if (wave == 0) {
      // A tile: 128 rows x 32 cols; pad 16DW interval (enc 3), 4DW amount (enc 3) -> pitch 40
      tdm_load_2d((unsigned)(size_t)&sA[buf][0], A + m0 * DMODEL + k0,
                  32u, 128u, (unsigned)DMODEL, 1u << 20, (unsigned)DMODEL, 3u, 3u, 1u);
      // B tile: 64 rows x 32 cols
      tdm_load_2d((unsigned)(size_t)&sB[buf][0], W + (size_t)n0 * DMODEL + k0,
                  32u, 64u, (unsigned)DMODEL, 1u << 20, (unsigned)DMODEL, 3u, 3u, 1u);
    }
  };
  issueTiles(0, 0);
  if (wave == 0) __builtin_amdgcn_s_wait_tensorcnt(0);
  __syncthreads();
#else
  auto loadTile = [&](int k0, int buf) {
    #pragma unroll
    for (int i = 0; i < 2; ++i) {
      int idx = tid + i * 256;
      int row = idx >> 2, seg = idx & 3;
      uint4 va = *(const uint4*)(A + (m0 + row) * DMODEL + k0 + seg * 8);
      *(uint4*)(&sA[buf][row * 40 + seg * 8]) = va;
    }
    {
      int row = tid >> 2, seg = tid & 3;
      uint4 vb = *(const uint4*)(W + (long)(n0 + row) * DMODEL + k0 + seg * 8);
      *(uint4*)(&sB[buf][row * 40 + seg * 8]) = vb;
    }
  };
  loadTile(0, 0);
  __syncthreads();
#endif

  const int rr = lane & 15;
  const int ka = (lane < 16) ? 0 : 8;   // A frag: K 0-7/16-23 vs 8-15/24-31
  const int kb = (lane < 16) ? 0 : 8;   // B frag: K 0-7/16-23 vs 8-15/24-31

  for (int kk = 0; kk < 16; ++kk) {
    int cur = kk & 1;
#if USE_TDM
    if (kk < 15) issueTiles((kk + 1) * 32, cur ^ 1);
#else
    if (kk < 15) loadTile((kk + 1) * 32, cur ^ 1);
#endif
    const bf16_t* pA = sA[cur];
    const bf16_t* pB = sB[cur];
    v16bf af[2], bfv[2];
    #pragma unroll
    for (int mt = 0; mt < 2; ++mt) {
      int m = wm * 32 + mt * 16 + rr;
      v8bf lo = *(const v8bf*)(pA + m * 40 + ka);
      v8bf hi = *(const v8bf*)(pA + m * 40 + ka + 16);
      af[mt] = cat8(lo, hi);
    }
    #pragma unroll
    for (int nt = 0; nt < 2; ++nt) {
      int n = wn * 32 + nt * 16 + rr;
      v8bf lo = *(const v8bf*)(pB + n * 40 + kb);
      v8bf hi = *(const v8bf*)(pB + n * 40 + kb + 16);
      bfv[nt] = cat8(lo, hi);
    }
    #pragma unroll
    for (int mt = 0; mt < 2; ++mt)
      #pragma unroll
      for (int nt = 0; nt < 2; ++nt)
        acc[mt][nt] = __builtin_amdgcn_wmma_f32_16x16x32_bf16(
            false, af[mt], false, bfv[nt], (short)0, acc[mt][nt], false, false);
#if USE_TDM
    if (wave == 0) __builtin_amdgcn_s_wait_tensorcnt(0);
#endif
    __syncthreads();
  }

  const int mh = (lane >> 4) * 8;
  if (mode != 2) {
    // repack through LDS (pitch 72) -> coalesced b128 stores
    bf16_t* sOut = &sA[0][0];   // 128*72*2 = 18432 B <= sizeof(sA)
    #pragma unroll
    for (int nt = 0; nt < 2; ++nt) {
      int nc = wn * 32 + nt * 16 + rr;
      float bv = bias[n0 + nc];
      #pragma unroll
      for (int mt = 0; mt < 2; ++mt) {
        int mb = wm * 32 + mt * 16 + mh;
        #pragma unroll
        for (int r = 0; r < 8; ++r) {
          float v = acc[mt][nt][r] + bv;
          if (mode == 1) v = (v > 0.0f) ? (v + 1.0f) : __expf(v);  // elu(v)+1
          sOut[(mb + r) * 72 + nc] = (bf16_t)v;
        }
      }
    }
    __syncthreads();
    #pragma unroll
    for (int i = 0; i < 4; ++i) {
      int idx = tid + i * 256;
      int row = idx >> 3, seg = idx & 7;
      *(uint4*)(outb + (m0 + row) * DMODEL + n0 + seg * 8) =
          *(const uint4*)(&sOut[row * 72 + seg * 8]);
    }
  } else {
    #pragma unroll
    for (int nt = 0; nt < 2; ++nt) {
      int n = n0 + wn * 32 + nt * 16 + rr;
      float bv = bias[n];
      #pragma unroll
      for (int mt = 0; mt < 2; ++mt) {
        long mb = m0 + wm * 32 + mt * 16 + mh;
        #pragma unroll
        for (int r = 0; r < 8; ++r) {
          long t = mb + r;
          outf[t * DMODEL + n] = acc[mt][nt][r] + bv + resid[t * DMODEL + n];
        }
      }
    }
  }
}

// ---------------------------------------------------------------- kv = K^T V partials + ksum partials
__global__ __launch_bounds__(256)
void kv_reduce(const bf16_t* __restrict__ Kb, const bf16_t* __restrict__ Vb,
               float* __restrict__ part)
{
  __shared__ bf16_t sk[32 * 72];
  __shared__ bf16_t sv[32 * 72];
  const int tid = threadIdx.x, lane = tid & 31, wave = tid >> 5;
  const int pair = blockIdx.x;   // b*8 + head
  const int chunk = blockIdx.y;
  const int b = pair >> 3, hh = pair & 7;
  const long t0 = (long)b * LTOK + (long)chunk * KV_CHUNK;
  const int col0 = hh * HDIM;

  v8f acc[2] = {};
  float kacc = 0.0f;
  const int s0 = wave * 2;
  const int mt = s0 >> 2;                 // shared by both subtiles of this wave
  const int nts0 = s0 & 3, nts1 = (s0 + 1) & 3;
  const int lrow = tid >> 3, lseg = tid & 7;
  const int rr = lane & 15;
  const int ka = (lane < 16) ? 0 : 8;
  const int kb = (lane < 16) ? 0 : 8;

  for (int it = 0; it < KV_CHUNK / 32; ++it) {
    long tb = t0 + it * 32;
#if USE_TDM
    if (wave == 0) {
      // 32 rows x 64 cols; pad 32DW interval (enc 4), 4DW amount (enc 3) -> pitch 72
      tdm_load_2d((unsigned)(size_t)&sk[0], Kb + tb * DMODEL + col0,
                  64u, 32u, (unsigned)DMODEL, 1u << 20, (unsigned)DMODEL, 4u, 3u, 1u);
      tdm_load_2d((unsigned)(size_t)&sv[0], Vb + tb * DMODEL + col0,
                  64u, 32u, (unsigned)DMODEL, 1u << 20, (unsigned)DMODEL, 4u, 3u, 1u);
      __builtin_amdgcn_s_wait_tensorcnt(0);
    }
#else
    *(uint4*)(&sk[lrow * 72 + lseg * 8]) =
        *(const uint4*)(Kb + (tb + lrow) * DMODEL + col0 + lseg * 8);
    *(uint4*)(&sv[lrow * 72 + lseg * 8]) =
        *(const uint4*)(Vb + (tb + lrow) * DMODEL + col0 + lseg * 8);
#endif
    __syncthreads();

    if (tid < 64) {
      float s = 0.0f;
      #pragma unroll
      for (int r = 0; r < 32; ++r) s += (float)sk[r * 72 + tid];
      kacc += s;
    }
    // A = K^T tile (transpose gathered from LDS)
    int m = mt * 16 + rr;
    v16bf a;
    #pragma unroll
    for (int j = 0; j < 8; ++j) {
      a[j]     = sk[(ka + j) * 72 + m];
      a[j + 8] = sk[(ka + 16 + j) * 72 + m];
    }
    #pragma unroll
    for (int i = 0; i < 2; ++i) {
      int nt = i ? nts1 : nts0;
      int n = nt * 16 + rr;
      v16bf bfr;
      #pragma unroll
      for (int j = 0; j < 16; ++j) bfr[j] = sv[(kb + (j & 7) + ((j >> 3) << 4)) * 72 + n];
      acc[i] = __builtin_amdgcn_wmma_f32_16x16x32_bf16(
          false, a, false, bfr, (short)0, acc[i], false, false);
    }
    __syncthreads();
  }

  float* pp = part + (size_t)(pair * NCHUNK + chunk) * PART_STRIDE;
  const int mh = (lane >> 4) * 8;
  #pragma unroll
  for (int i = 0; i < 2; ++i) {
    int nt = i ? nts1 : nts0;
    int n = nt * 16 + rr;
    int mb = mt * 16 + mh;
    #pragma unroll
    for (int r = 0; r < 8; ++r) pp[(mb + r) * 64 + n] = acc[i][r];
  }
  if (tid < 64) pp[4096 + tid] = kacc;
}

// ---------------------------------------------------------------- reduce partials -> kv(bf16), ksum(f32)
__global__ __launch_bounds__(256)
void kv_finalize(const float* __restrict__ part, bf16_t* __restrict__ kvb,
                 float* __restrict__ ksum)
{
  const int pair = blockIdx.x, tid = threadIdx.x;
  const float* base = part + (size_t)pair * NCHUNK * PART_STRIDE;
  for (int e = tid; e < 4096; e += 256) {
    float s = 0.0f;
    #pragma unroll
    for (int c = 0; c < NCHUNK; ++c) s += base[c * PART_STRIDE + e];
    kvb[pair * 4096 + e] = (bf16_t)s;
  }
  if (tid < 64) {
    float s = 0.0f;
    #pragma unroll
    for (int c = 0; c < NCHUNK; ++c) s += base[c * PART_STRIDE + 4096 + tid];
    ksum[pair * 64 + tid] = s;
  }
}

// ---------------------------------------------------------------- out = (q @ kv) / (q . ksum + eps)
__global__ __launch_bounds__(256)
void attn_apply(const bf16_t* __restrict__ Qb, const bf16_t* __restrict__ kvb,
                const float* __restrict__ ksum, bf16_t* __restrict__ outb)
{
  __shared__ float snorm[NHEADS][16];
#if USE_TDM
  __shared__ bf16_t skv[NHEADS][64 * 64];
#endif
  const int tid = threadIdx.x, lane = tid & 31, head = tid >> 5;
  const long t0 = (long)blockIdx.x * 16;
  const int b = (int)(t0 / LTOK);
  const int pair = b * NHEADS + head;
  const int col0 = head * HDIM;
  const int rr = lane & 15;
  const int ka = (lane < 16) ? 0 : 8;
  const int kb = (lane < 16) ? 0 : 8;

#if USE_TDM
  // each wave DMAs its head's 64x64 kv tile into LDS (contiguous, no pad)
  tdm_load_2d((unsigned)(size_t)&skv[head][0], kvb + (size_t)pair * 4096,
              64u, 64u, 64u, 1u << 20, 64u, 0u, 0u, 0u);
  __builtin_amdgcn_s_wait_tensorcnt(0);
#endif

  if (lane < 16) {
    long t = t0 + lane;
    float s = 0.0f;
    #pragma unroll
    for (int d = 0; d < HDIM; ++d)
      s += (float)Qb[t * DMODEL + col0 + d] * ksum[pair * 64 + d];
    snorm[head][lane] = s;
  }
  __syncthreads();   // fence: publish TDM-written LDS + snorm

  v16bf a[2];
  #pragma unroll
  for (int ks = 0; ks < 2; ++ks) {
    const bf16_t* p = Qb + (t0 + rr) * DMODEL + col0 + ks * 32 + ka;
    v8bf lo = *(const v8bf*)p;
    v8bf hi = *(const v8bf*)(p + 16);
    a[ks] = cat8(lo, hi);
  }

  v8f acc[4] = {};
  #pragma unroll
  for (int nt = 0; nt < 4; ++nt) {
    #pragma unroll
    for (int ks = 0; ks < 2; ++ks) {
      int n = nt * 16 + rr;
      v16bf bfr;
      #pragma unroll
      for (int j = 0; j < 16; ++j) {
#if USE_TDM
        bfr[j] = skv[head][(ks * 32 + kb + (j & 7) + ((j >> 3) << 4)) * 64 + n];
#else
        bfr[j] = kvb[pair * 4096 + (ks * 32 + kb + (j & 7) + ((j >> 3) << 4)) * 64 + n];
#endif
      }
      acc[nt] = __builtin_amdgcn_wmma_f32_16x16x32_bf16(
          false, a[ks], false, bfr, (short)0, acc[nt], false, false);
    }
  }

  const int mh = (lane >> 4) * 8;
  #pragma unroll
  for (int nt = 0; nt < 4; ++nt) {
    #pragma unroll
    for (int r = 0; r < 8; ++r) {
      long t = t0 + mh + r;
      float nv = snorm[head][mh + r] + 1e-6f;
      outb[t * DMODEL + col0 + nt * 16 + rr] = (bf16_t)(acc[nt][r] / nv);
    }
  }
}

// ---------------------------------------------------------------- host
extern "C" void kernel_launch(void* const* d_in, const int* in_sizes, int n_in,
                              void* d_out, int out_size, void* d_ws, size_t ws_size,
                              hipStream_t stream)
{
  const float* x    = (const float*)d_in[0];
  const float* Wq   = (const float*)d_in[1];
  const float* bq   = (const float*)d_in[2];
  const float* Wk   = (const float*)d_in[3];
  const float* bk   = (const float*)d_in[4];
  const float* Wv   = (const float*)d_in[5];
  const float* bv   = (const float*)d_in[6];
  const float* Wo   = (const float*)d_in[7];
  const float* bo   = (const float*)d_in[8];
  const float* ln_g = (const float*)d_in[9];
  const float* ln_b = (const float*)d_in[10];
  float* out = (float*)d_out;

  const size_t WBYTES = (size_t)DMODEL * DMODEL * sizeof(bf16_t);       // 512 KB
  const size_t TBYTES = (size_t)TOKENS * DMODEL * sizeof(bf16_t);       // 96 MB
  uint8_t* ws = (uint8_t*)d_ws;
  bf16_t* wq_bf = (bf16_t*)(ws);
  bf16_t* wk_bf = (bf16_t*)(ws + WBYTES);
  bf16_t* wv_bf = (bf16_t*)(ws + 2 * WBYTES);
  bf16_t* wo_bf = (bf16_t*)(ws + 3 * WBYTES);
  bf16_t* h_bf  = (bf16_t*)(ws + 4 * WBYTES);
  bf16_t* q_bf  = (bf16_t*)(ws + 4 * WBYTES + TBYTES);
  bf16_t* k_bf  = (bf16_t*)(ws + 4 * WBYTES + 2 * TBYTES);
  bf16_t* v_bf  = (bf16_t*)(ws + 4 * WBYTES + 3 * TBYTES);
  float*  part  = (float*)(ws + 4 * WBYTES + 4 * TBYTES);               // 32*24*4160 f32
  bf16_t* kv_bf = (bf16_t*)((uint8_t*)part + (size_t)32 * NCHUNK * PART_STRIDE * 4);
  float*  ksum  = (float*)((uint8_t*)kv_bf + (size_t)32 * 4096 * sizeof(bf16_t));
  bf16_t* attn_bf = h_bf;  // h no longer needed after QKV projections

  const int wn = DMODEL * DMODEL;
  cvt_bf16<<<(wn + 255) / 256, 256, 0, stream>>>(Wq, wq_bf, wn);
  cvt_bf16<<<(wn + 255) / 256, 256, 0, stream>>>(Wk, wk_bf, wn);
  cvt_bf16<<<(wn + 255) / 256, 256, 0, stream>>>(Wv, wv_bf, wn);
  cvt_bf16<<<(wn + 255) / 256, 256, 0, stream>>>(Wo, wo_bf, wn);

  ln_kernel<<<TOKENS, 256, 0, stream>>>(x, ln_g, ln_b, h_bf);

  dim3 gg(TOKENS / 128, DMODEL / 64);
  gemm_bf16<<<gg, 256, 0, stream>>>(h_bf, wq_bf, bq, nullptr, q_bf, nullptr, 1);
  gemm_bf16<<<gg, 256, 0, stream>>>(h_bf, wk_bf, bk, nullptr, k_bf, nullptr, 1);
  gemm_bf16<<<gg, 256, 0, stream>>>(h_bf, wv_bf, bv, nullptr, v_bf, nullptr, 0);

  kv_reduce<<<dim3(32, NCHUNK), 256, 0, stream>>>(k_bf, v_bf, part);
  kv_finalize<<<32, 256, 0, stream>>>(part, kv_bf, ksum);

  attn_apply<<<TOKENS / 16, 256, 0, stream>>>(q_bf, kv_bf, ksum, attn_bf);

  gemm_bf16<<<gg, 256, 0, stream>>>(attn_bf, wo_bf, bo, x, nullptr, out, 2);
}
